// TokenDiscrepancyLoss_23759759082040
// MI455X (gfx1250) — matrix operand order repaired and checked by reading, not compiled, not run on hardware
//
#include <hip/hip_runtime.h>
#include <hip/hip_bf16.h>

// ---------------------------------------------------------------------------
// TokenDiscrepancyLoss fused kernel for gfx1250 (MI455X)
//   loss = 0.1 * sum_masked_n [ sum_k softmax(x W^T + b)[n,k] * mse(t_n, c_k) ]
//   mse(t,c) = (||t||^2 - 2 t.c + ||c||^2) / D
// bf16 WMMA dual-GEMM (logits + cross) + online softmax; [N,K] never
// materialized. NT=2 column tiles per wave iteration so each A fragment
// (ds_load) feeds 2x v_wmma -> fewer load-issue slots per matrix op.
// ---------------------------------------------------------------------------

typedef __attribute__((ext_vector_type(16))) __bf16 v16bf;
typedef __attribute__((ext_vector_type(8)))  __bf16 v8bf;
typedef __attribute__((ext_vector_type(8)))  float  v8f;

#define HD      1024          // H == D (fixed by problem shapes)
#define KSPLIT  4             // grid.y split of K
#define WAVES   4             // waves per workgroup (wave32)
#define TPB     (WAVES * 32)
#define NEG_BIG (-3.0e38f)

// ---------------- prep: f32 -> bf16 (round to nearest even) ----------------
__global__ __launch_bounds__(256) void tdl_cvt_bf16(const float* __restrict__ src,
                                                    unsigned short* __restrict__ dst,
                                                    size_t n) {
  size_t i = (size_t)blockIdx.x * blockDim.x + threadIdx.x;
  size_t stride = (size_t)gridDim.x * blockDim.x;
  for (; i < n; i += stride) {
    unsigned int u = __builtin_bit_cast(unsigned int, src[i]);
    unsigned int r = (u + 0x7FFFu + ((u >> 16) & 1u)) >> 16;
    dst[i] = (unsigned short)r;
  }
}

// ---------------- prep: per-row sum of squares (t_sq, c_sq) ----------------
__global__ __launch_bounds__(256) void tdl_rowsq(const float* __restrict__ src,
                                                 float* __restrict__ dst, int cols) {
  __shared__ float red[256];
  const float* row = src + (size_t)blockIdx.x * cols;
  float s = 0.f;
  for (int i = threadIdx.x; i < cols; i += 256) { float v = row[i]; s += v * v; }
  red[threadIdx.x] = s;
  __syncthreads();
  for (int o = 128; o > 0; o >>= 1) {
    if ((int)threadIdx.x < o) red[threadIdx.x] += red[threadIdx.x + o];
    __syncthreads();
  }
  if (threadIdx.x == 0) dst[blockIdx.x] = red[0];
}

// ----------------------------- A/B tile loads ------------------------------
// 16-bit A 16x32 fragment in this code's contiguous k order: lane L holds row M = L&15.
//   lanes 0-15 : halves 0..15 -> K=h+0..15
//   lanes 16-31: halves 0..15 -> K=h+16..31 (contiguous, matches B)
__device__ __forceinline__ v16bf tdl_load_a(const unsigned short* p, int koff) {
  v8bf lo = *(const v8bf*)(p + koff);        // ds_load_b128
  v8bf hi = *(const v8bf*)(p + 8 + koff);    // ds_load_b128
  return __builtin_shufflevector(lo, hi, 0, 1, 2, 3, 4, 5, 6, 7,
                                 8, 9, 10, 11, 12, 13, 14, 15);
}
// 16-bit B 32x16 layout: lane L holds col N = L&15;
//   lanes 0-15 hold K=h+0..15, lanes 16-31 hold K=h+16..31 (contiguous 32B).
__device__ __forceinline__ v16bf tdl_load_b(const unsigned short* p) {
  return *(const v16bf*)p;                   // 2x global_load_b128
}

// ------------------------------ fused kernel -------------------------------
// grid = (N/16, KSPLIT), block = 128. WG owns 16 token rows (x,t staged in
// LDS); each wave covers (K/KSPLIT/WAVES) codebook columns, 2 column tiles
// per iteration. C/D layout: lane holds col = lane&15, rows 8*half + i.
__global__ __launch_bounds__(TPB) void tdl_fused(
    const unsigned short* __restrict__ XB, const unsigned short* __restrict__ TB,
    const unsigned short* __restrict__ WB, const unsigned short* __restrict__ CB,
    const float* __restrict__ bproj, const float* __restrict__ tsq,
    const float* __restrict__ csq, float* __restrict__ partial, int K) {
  __shared__ unsigned short xlds[16 * HD];
  __shared__ unsigned short tlds[16 * HD];
  __shared__ float mm_s[WAVES][16];
  __shared__ float zz_s[WAVES][16];
  __shared__ float ss_s[WAVES][16];

  const int tid  = threadIdx.x;
  const int wave = tid >> 5;
  const int lane = tid & 31;
  const int half = lane >> 4;   // which 16-lane half of the wave
  const int ln   = lane & 15;
  const int r0   = blockIdx.x * 16;

  // Stage the 16 x-rows and 16 t-rows (bf16) into LDS.
  {
    const uint4* gx = (const uint4*)(XB + (size_t)r0 * HD);
    const uint4* gt = (const uint4*)(TB + (size_t)r0 * HD);
    uint4* lx = (uint4*)xlds;
    uint4* lt = (uint4*)tlds;
    const int nvec = 16 * HD / 8;
    for (int i = tid; i < nvec; i += TPB) { lx[i] = gx[i]; lt[i] = gt[i]; }
  }
  __syncthreads();

  float m[8], Z[8], S[8], tq[8];
#pragma unroll
  for (int i = 0; i < 8; ++i) { m[i] = NEG_BIG; Z[i] = 0.f; S[i] = 0.f; }
#pragma unroll
  for (int i = 0; i < 8; ++i) tq[i] = tsq[r0 + half * 8 + i];

  const int colsPerWave = (K / KSPLIT) / WAVES;        // 1024
  const int cb0 = blockIdx.y * (K / KSPLIT) + wave * colsPerWave;
  const int niter = colsPerWave / 32;                  // 32 (2 tiles / iter)
  const int koff = half * 16;
  const unsigned short* xrow = xlds + ln * HD;
  const unsigned short* trow = tlds + ln * HD;

  for (int t = 0; t < niter; ++t) {
    const int col0 = cb0 + t * 32 + ln;                // column tile 0
    const int col1 = col0 + 16;                        // column tile 1
    const unsigned short* wrow0 = WB + (size_t)col0 * HD + half * 16;
    const unsigned short* crow0 = CB + (size_t)col0 * HD + half * 16;
    const unsigned short* wrow1 = WB + (size_t)col1 * HD + half * 16;
    const unsigned short* crow1 = CB + (size_t)col1 * HD + half * 16;
    if (t + 1 < niter) {                               // global_prefetch_b8
      __builtin_prefetch(wrow0 + (size_t)32 * HD, 0, 3);
      __builtin_prefetch(crow0 + (size_t)32 * HD, 0, 3);
      __builtin_prefetch(wrow1 + (size_t)32 * HD, 0, 3);
      __builtin_prefetch(crow1 + (size_t)32 * HD, 0, 3);
    }

    v8f accL0 = {0.f, 0.f, 0.f, 0.f, 0.f, 0.f, 0.f, 0.f};
    v8f accC0 = {0.f, 0.f, 0.f, 0.f, 0.f, 0.f, 0.f, 0.f};
    v8f accL1 = {0.f, 0.f, 0.f, 0.f, 0.f, 0.f, 0.f, 0.f};
    v8f accC1 = {0.f, 0.f, 0.f, 0.f, 0.f, 0.f, 0.f, 0.f};
#pragma unroll 2
    for (int h = 0; h < HD; h += 32) {
      v16bf ax  = tdl_load_a(xrow + h, koff);          // shared by both tiles
      v16bf at  = tdl_load_a(trow + h, koff);
      v16bf bw0 = tdl_load_b(wrow0 + h);
      v16bf bc0 = tdl_load_b(crow0 + h);
      v16bf bw1 = tdl_load_b(wrow1 + h);
      v16bf bc1 = tdl_load_b(crow1 + h);
      accL0 = __builtin_amdgcn_wmma_f32_16x16x32_bf16(false, ax, false, bw0,
                                                      (short)0, accL0, false, false);
      accC0 = __builtin_amdgcn_wmma_f32_16x16x32_bf16(false, at, false, bc0,
                                                      (short)0, accC0, false, false);
      accL1 = __builtin_amdgcn_wmma_f32_16x16x32_bf16(false, ax, false, bw1,
                                                      (short)0, accL1, false, false);
      accC1 = __builtin_amdgcn_wmma_f32_16x16x32_bf16(false, at, false, bc1,
                                                      (short)0, accC1, false, false);
    }

    const float bq0 = bproj[col0];
    const float cq0 = csq[col0];
    const float bq1 = bproj[col1];
    const float cq1 = csq[col1];
#pragma unroll
    for (int i = 0; i < 8; ++i) {
      float l0   = accL0[i] + bq0;
      float l1   = accL1[i] + bq1;
      float mse0 = (tq[i] - 2.f * accC0[i] + cq0) * (1.f / (float)HD);
      float mse1 = (tq[i] - 2.f * accC1[i] + cq1) * (1.f / (float)HD);
      float nm   = fmaxf(m[i], fmaxf(l0, l1));
      float eo   = __expf(m[i] - nm);
      float e0   = __expf(l0 - nm);
      float e1   = __expf(l1 - nm);
      Z[i] = Z[i] * eo + e0 + e1;
      S[i] = S[i] * eo + e0 * mse0 + e1 * mse1;
      m[i] = nm;
    }
  }

  // Log-sum-exp merge across the 16 lanes of each half (columns 0..15).
#pragma unroll
  for (int mask = 1; mask < 16; mask <<= 1) {
#pragma unroll
    for (int i = 0; i < 8; ++i) {
      float m2 = __shfl_xor(m[i], mask, 32);
      float z2 = __shfl_xor(Z[i], mask, 32);
      float s2 = __shfl_xor(S[i], mask, 32);
      float nm = fmaxf(m[i], m2);
      float e1 = __expf(m[i] - nm);
      float e2 = __expf(m2 - nm);
      Z[i] = Z[i] * e1 + z2 * e2;
      S[i] = S[i] * e1 + s2 * e2;
      m[i] = nm;
    }
  }

  if (ln == 0) {
#pragma unroll
    for (int i = 0; i < 8; ++i) {
      int r = half * 8 + i;
      mm_s[wave][r] = m[i]; zz_s[wave][r] = Z[i]; ss_s[wave][r] = S[i];
    }
  }
  __syncthreads();

  if (tid < 16) {  // merge the 4 waves' column ranges, emit one partial/row
    float mm = NEG_BIG, zz = 0.f, ss = 0.f;
#pragma unroll
    for (int w = 0; w < WAVES; ++w) {
      float m2 = mm_s[w][tid], z2 = zz_s[w][tid], s2 = ss_s[w][tid];
      float nm = fmaxf(mm, m2);
      float e1 = __expf(mm - nm);
      float e2 = __expf(m2 - nm);
      zz = zz * e1 + z2 * e2;
      ss = ss * e1 + s2 * e2;
      mm = nm;
    }
    size_t idx = ((size_t)(r0 + tid) * KSPLIT + blockIdx.y) * 3;
    partial[idx + 0] = mm;
    partial[idx + 1] = zz;
    partial[idx + 2] = ss;
  }
}

// --------------------------- final reduction -------------------------------
__global__ __launch_bounds__(256) void tdl_final(const float* __restrict__ partial,
                                                 const int* __restrict__ tt,
                                                 float* __restrict__ out, int N) {
  __shared__ float red[256];
  float acc = 0.f;
  for (int r = threadIdx.x; r < N; r += 256) {
    float mm = NEG_BIG, zz = 0.f, ss = 0.f;
    for (int s = 0; s < KSPLIT; ++s) {
      const float* p = partial + ((size_t)r * KSPLIT + s) * 3;
      float m2 = p[0], z2 = p[1], s2 = p[2];
      float nm = fmaxf(mm, m2);
      float e1 = __expf(mm - nm);
      float e2 = __expf(m2 - nm);
      zz = zz * e1 + z2 * e2;
      ss = ss * e1 + s2 * e2;
      mm = nm;
    }
    if (tt[r] == 1) acc += ss / zz;
  }
  red[threadIdx.x] = acc;
  __syncthreads();
  for (int o = 128; o > 0; o >>= 1) {
    if ((int)threadIdx.x < o) red[threadIdx.x] += red[threadIdx.x + o];
    __syncthreads();
  }
  if (threadIdx.x == 0) out[0] = red[0] * 0.1f;
}

// ------------------------------- launcher ----------------------------------
extern "C" void kernel_launch(void* const* d_in, const int* in_sizes, int n_in,
                              void* d_out, int out_size, void* d_ws, size_t ws_size,
                              hipStream_t stream) {
  const float* x  = (const float*)d_in[0];  // [B,S,H]
  const int*   tt = (const int*)d_in[1];    // [B,S]
  const float* t  = (const float*)d_in[2];  // [B,S,H]
  const float* W  = (const float*)d_in[3];  // [K,H]
  const float* b  = (const float*)d_in[4];  // [K]
  const float* C  = (const float*)d_in[5];  // [K,H]

  const int N = in_sizes[1];                // B*S = 4096
  const int K = in_sizes[4];                // 16384
  // H fixed at HD (=1024) by the problem shapes.

  char* ws = (char*)d_ws;
  auto carve = [&](size_t bytes) {
    char* p = ws;
    ws += (bytes + 255) & ~(size_t)255;
    return p;
  };
  unsigned short* XB = (unsigned short*)carve((size_t)N * HD * 2);
  unsigned short* TB = (unsigned short*)carve((size_t)N * HD * 2);
  unsigned short* WB = (unsigned short*)carve((size_t)K * HD * 2);
  unsigned short* CB = (unsigned short*)carve((size_t)K * HD * 2);
  float* tsq     = (float*)carve((size_t)N * 4);
  float* csq     = (float*)carve((size_t)K * 4);
  float* partial = (float*)carve((size_t)N * KSPLIT * 3 * 4);

  // f32 -> bf16 conversions
  tdl_cvt_bf16<<<2048, 256, 0, stream>>>(x, XB, (size_t)N * HD);
  tdl_cvt_bf16<<<2048, 256, 0, stream>>>(t, TB, (size_t)N * HD);
  tdl_cvt_bf16<<<4096, 256, 0, stream>>>(W, WB, (size_t)K * HD);
  tdl_cvt_bf16<<<4096, 256, 0, stream>>>(C, CB, (size_t)K * HD);
  // row sums of squares
  tdl_rowsq<<<N, 256, 0, stream>>>(t, tsq, HD);
  tdl_rowsq<<<K, 256, 0, stream>>>(C, csq, HD);
  // fused dual-GEMM + online softmax
  dim3 grid(N / 16, KSPLIT);
  tdl_fused<<<grid, TPB, 0, stream>>>(XB, TB, WB, CB, b, tsq, csq, partial, K);
  // merge slices, mask, scalar reduce
  tdl_final<<<1, 256, 0, stream>>>(partial, tt, (float*)d_out, N);
}
